// CPIASA_17248588661158
// MI455X (gfx1250) — matrix operand order repaired and checked by reading, not compiled, not run on hardware
//
#include <hip/hip_runtime.h>
#include <math.h>

// ---------------------------------------------------------------------------
// CDNA5 (gfx1250) implementation. wave32. All GEMM-shaped work goes through a
// single bf16 WMMA kernel (v_wmma_f32_16x16x32_bf16). The 2D FFT/iFFT is done
// as DFT-by-GEMM with a precomputed 192x192 cos/sin matrix, so it also runs on
// the matrix pipes.
// ---------------------------------------------------------------------------

typedef __attribute__((ext_vector_type(2)))  __bf16 v2bf;
typedef __attribute__((ext_vector_type(8)))  __bf16 v8bf;
typedef __attribute__((ext_vector_type(16))) __bf16 v16bf;
typedef __attribute__((ext_vector_type(8)))  float  v8f;

static constexpr int  HWD = 192;
static constexpr long P   = 36864;          // 192*192 pixels per (b,c) image
static constexpr int  NB  = 4;              // batch

__device__ __forceinline__ float gelu_exact(float x) {
  return 0.5f * x * (1.0f + erff(x * 0.70710678118654752440f));
}
__device__ __forceinline__ float sigm(float x) {
  return 1.0f / (1.0f + expf(-x));
}
__device__ __forceinline__ v8bf cvt8(float4 a, float4 b) {
  v8bf r;
  r[0] = (__bf16)a.x; r[1] = (__bf16)a.y; r[2] = (__bf16)a.z; r[3] = (__bf16)a.w;
  r[4] = (__bf16)b.x; r[5] = (__bf16)b.y; r[6] = (__bf16)b.z; r[7] = (__bf16)b.w;
  return r;
}

// ---------------------------------------------------------------------------
// Generic bf16 WMMA GEMM:  D = accum*D + alpha*(A x B) (+bias[row]) (+act)
//   A[m,k]  fp32, row stride lda, batch base = zo*a_so + zi*a_si
//   B[k,n]  fp32, element at k*b_sk + n*b_sn, batch base = zo*b_so + zi*b_si
//   D[m,n]  fp32, row stride ldd, batch base = zo*d_so + zi*d_si
// grid: (ceil(N/64), ceil(M/32), Z), block 256 = 8 waves, wave tile 16x16,
// K stepped by 64 (2 WMMAs per LDS stage). z -> zo = z/ZH, zi = z%ZH.
// LDS: A tile k-major [m][k]; B tile stored TRANSPOSED [n][k] so each lane's
// fragment is two aligned ds_load_b128 per matrix.
// ---------------------------------------------------------------------------
__global__ __launch_bounds__(256) void gemm_bf16_wmma(
    const float* __restrict__ A, int lda, long a_so, long a_si,
    const float* __restrict__ B, long b_sk, long b_sn, long b_so, long b_si,
    float* D, int ldd, long d_so, long d_si,
    const float* __restrict__ bias,
    int M, int N, int K, int ZH, float alpha, int accum, int act)
{
  __shared__ __bf16 As[32][72];   // [m][k], row = 144 B (16B aligned)
  __shared__ __bf16 Bs[64][72];   // [n][k], transposed

  const int z  = blockIdx.z;
  const int zo = z / ZH, zi = z % ZH;
  const float* Ab = A + (size_t)zo * a_so + (size_t)zi * a_si;
  const float* Bb = B + (size_t)zo * b_so + (size_t)zi * b_si;
  float*       Db = D + (size_t)zo * d_so + (size_t)zi * d_si;

  const int m0 = blockIdx.y * 32;
  const int n0 = blockIdx.x * 64;
  const int t    = threadIdx.x;
  const int lane = t & 31;          // wave32
  const int wave = t >> 5;
  const int half = lane >> 4;
  const int l16  = lane & 15;
  const int wm   = wave >> 2;       // 0..1
  const int wn   = wave & 3;        // 0..3

  v8f acc;
#pragma unroll
  for (int i = 0; i < 8; ++i) acc[i] = 0.0f;

  for (int k0 = 0; k0 < K; k0 += 64) {
    const bool kfull = (k0 + 64 <= K);

    // ---- stage A tile (32 x 64) ----
    if ((m0 + 32 <= M) && kfull) {           // uniform branch: fast path
      const int mm = t >> 3, kk = (t & 7) * 8;
      const float* ap = Ab + (long)(m0 + mm) * lda + (k0 + kk);
      float4 a0 = *(const float4*)ap;
      float4 a1 = *(const float4*)(ap + 4);
      *(v8bf*)&As[mm][kk] = cvt8(a0, a1);
    } else {
#pragma unroll
      for (int j = 0; j < 8; ++j) {
        int e = t + j * 256;
        int mm = e >> 6, kk = e & 63;
        float v = 0.0f;
        if (m0 + mm < M && k0 + kk < K)
          v = Ab[(long)(m0 + mm) * lda + (k0 + kk)];
        As[mm][kk] = (__bf16)v;
      }
    }

    // ---- stage B tile (64k x 64n) into Bs[n][k] ----
    if (b_sn == 1 && kfull && (n0 + 64 <= N)) {
      // n is contiguous in memory: each thread takes 2 k-rows x 8 n
      const int kk = (t & 31) * 2, nn0 = (t >> 5) * 8;
      const float* bp = Bb + (long)(k0 + kk) * b_sk + (n0 + nn0);
      float4 x00 = *(const float4*)bp;       float4 x01 = *(const float4*)(bp + 4);
      const float* bq = bp + b_sk;
      float4 x10 = *(const float4*)bq;       float4 x11 = *(const float4*)(bq + 4);
      float r0[8] = {x00.x, x00.y, x00.z, x00.w, x01.x, x01.y, x01.z, x01.w};
      float r1[8] = {x10.x, x10.y, x10.z, x10.w, x11.x, x11.y, x11.z, x11.w};
#pragma unroll
      for (int j = 0; j < 8; ++j) {
        v2bf p; p[0] = (__bf16)r0[j]; p[1] = (__bf16)r1[j];
        *(v2bf*)&Bs[nn0 + j][kk] = p;
      }
      if (k0 + 64 < K) __builtin_prefetch(bp + 64 * b_sk, 0, 1);   // next panel
    } else if (b_sk == 1 && kfull && (n0 + 64 <= N)) {
      // k is contiguous in memory: each thread takes 1 n x 16 k
      const int nn = t >> 2, kk = (t & 3) * 16;
      const float* bp = Bb + (long)(k0 + kk) + (long)(n0 + nn) * b_sn;
      float4 c0 = *(const float4*)bp;        float4 c1 = *(const float4*)(bp + 4);
      float4 c2 = *(const float4*)(bp + 8);  float4 c3 = *(const float4*)(bp + 12);
      *(v8bf*)&Bs[nn][kk]     = cvt8(c0, c1);
      *(v8bf*)&Bs[nn][kk + 8] = cvt8(c2, c3);
      if (k0 + 64 < K) __builtin_prefetch(bp + 64, 0, 1);
    } else {
#pragma unroll
      for (int j = 0; j < 16; ++j) {
        int e = t + j * 256;
        int kk = e >> 6, nn = e & 63;
        float v = 0.0f;
        if (k0 + kk < K && n0 + nn < N)
          v = Bb[(long)(k0 + kk) * b_sk + (long)(n0 + nn) * b_sn];
        Bs[nn][kk] = (__bf16)v;
      }
    }
    __syncthreads();

    // ---- two WMMAs per stage; fragments = aligned ds_load_b128 pairs ----
    const int ar = wm * 16 + l16;
    const int bc = wn * 16 + l16;
#pragma unroll
    for (int s = 0; s < 2; ++s) {
      const int ks = s * 32;
      v8bf alo = *(const v8bf*)&As[ar][ks + half * 8];
      v8bf ahi = *(const v8bf*)&As[ar][ks + 16 + half * 8];
      v8bf blo = *(const v8bf*)&Bs[bc][ks + half * 8];
      v8bf bhi = *(const v8bf*)&Bs[bc][ks + 16 + half * 8];
      v16bf af = __builtin_shufflevector(alo, ahi,
          0, 1, 2, 3, 4, 5, 6, 7, 8, 9, 10, 11, 12, 13, 14, 15);
      v16bf bf = __builtin_shufflevector(blo, bhi,
          0, 1, 2, 3, 4, 5, 6, 7, 8, 9, 10, 11, 12, 13, 14, 15);
      acc = __builtin_amdgcn_wmma_f32_16x16x32_bf16(
          false, af, false, bf, (short)0, acc, false, false);
    }
    __syncthreads();
  }

  // Epilogue: C/D layout -> row = half*8 + v, col = lane%16
#pragma unroll
  for (int v = 0; v < 8; ++v) {
    int row = m0 + wm * 16 + half * 8 + v;
    int col = n0 + wn * 16 + l16;
    if (row < M && col < N) {
      long idx = (long)row * ldd + col;
      float val = acc[v] * alpha;
      if (accum) val += Db[idx];
      if (bias)  val += bias[row];
      if (act == 1)      val = fmaxf(val, 0.0f);
      else if (act == 2) val = sigm(val);
      Db[idx] = val;
    }
  }
}

// ---------------------------------------------------------------------------
// Depthwise 3x3 conv, pad=1.  in channel = oc / ratio.
// ---------------------------------------------------------------------------
__global__ __launch_bounds__(256) void dwconv3x3_k(
    const float* __restrict__ in, long in_bstride,
    const float* __restrict__ w9, const float* __restrict__ bias,
    float* __restrict__ out, int Cout, int ratio)
{
  size_t i = (size_t)blockIdx.x * 256 + threadIdx.x;
  size_t total = (size_t)NB * Cout * P;
  if (i >= total) return;
  int  b  = (int)(i / ((size_t)Cout * P));
  int  oc = (int)((i / P) % Cout);
  long p  = (long)(i % P);
  int  h  = (int)(p / HWD), w = (int)(p % HWD);
  int  ic = oc / ratio;
  const float* src = in + (size_t)b * in_bstride + (size_t)ic * P;
  float acc = bias[oc];
#pragma unroll
  for (int dy = -1; dy <= 1; ++dy) {
    int hh = h + dy; if (hh < 0 || hh >= HWD) continue;
#pragma unroll
    for (int dx = -1; dx <= 1; ++dx) {
      int ww = w + dx; if (ww < 0 || ww >= HWD) continue;
      acc += w9[oc * 9 + (dy + 1) * 3 + (dx + 1)] * src[(long)hh * HWD + ww];
    }
  }
  out[(size_t)b * Cout * P + (size_t)oc * P + p] = acc;
}

// ---------------------------------------------------------------------------
// Per-row sum / sum-of-squares.  row rid = o*R + r, base = o*So + r*Sr.
// ---------------------------------------------------------------------------
__global__ __launch_bounds__(256) void row_reduce_k(
    const float* __restrict__ in, long So, long Sr, int R, long L,
    float* __restrict__ osum, float* __restrict__ osq)
{
  int rid = blockIdx.x;
  int o = rid / R, r = rid % R;
  const float* ptr = in + (size_t)o * So + (size_t)r * Sr;
  float s = 0.0f, q = 0.0f;
  for (long j = threadIdx.x; j < L; j += 256) { float v = ptr[j]; s += v; q += v * v; }
  __shared__ float sh[256], sh2[256];
  sh[threadIdx.x] = s; sh2[threadIdx.x] = q;
  __syncthreads();
  for (int st = 128; st > 0; st >>= 1) {
    if ((int)threadIdx.x < st) { sh[threadIdx.x] += sh[threadIdx.x + st];
                                 sh2[threadIdx.x] += sh2[threadIdx.x + st]; }
    __syncthreads();
  }
  if (threadIdx.x == 0) { osum[rid] = sh[0]; osq[rid] = sh2[0]; }
}

// softmax over 256 channel means, per batch
__global__ __launch_bounds__(256) void spr_softmax_k(
    const float* __restrict__ rsum, float* __restrict__ ssoft)
{
  int b = blockIdx.x, c = threadIdx.x;
  __shared__ float sh[256];
  float m = rsum[b * 256 + c] / (float)P;
  sh[c] = m; __syncthreads();
  for (int s = 128; s > 0; s >>= 1) { if (c < s) sh[c] = fmaxf(sh[c], sh[c + s]); __syncthreads(); }
  float mx = sh[0]; __syncthreads();
  float e = expf(m - mx);
  sh[c] = e; __syncthreads();
  for (int s = 128; s > 0; s >>= 1) { if (c < s) sh[c] += sh[c + s]; __syncthreads(); }
  ssoft[b * 256 + c] = e / sh[0];
}

// h2 = gelu(s[b,c] * h2)
__global__ __launch_bounds__(256) void scale_gelu_k(
    float* __restrict__ h2, const float* __restrict__ ssoft)
{
  size_t i = (size_t)blockIdx.x * 256 + threadIdx.x;
  if (i >= (size_t)NB * 256 * P) return;
  int b = (int)(i / (256 * P));
  int c = (int)((i / P) % 256);
  h2[i] = gelu_exact(ssoft[b * 256 + c] * h2[i]);
}

// q,k L2-normalize along n (rsq holds sum of squares per (b, ch 0..255))
__global__ __launch_bounds__(256) void qk_norm_k(
    float* __restrict__ qkvd, const float* __restrict__ rsq)
{
  size_t i = (size_t)blockIdx.x * 256 + threadIdx.x;
  if (i >= (size_t)NB * 256 * P) return;
  int  b  = (int)(i / (256 * P));
  int  ch = (int)((i / P) % 256);
  long p  = (long)(i % P);
  float nrm = sqrtf(rsq[b * 256 + ch]);
  float sc  = 1.0f / fmaxf(nrm, 1e-12f);
  qkvd[(size_t)b * 384 * P + (size_t)ch * P + p] *= sc;
}

// gv -> dyn_k, plus attn scale sum
__global__ void gv_dynk_k(const float* rsum,
                          const float* a1, const float* a2,
                          const float* a3, const float* a4, float* scal)
{
  if (threadIdx.x == 0 && blockIdx.x == 0) {
    float s  = rsum[0] + rsum[1] + rsum[2] + rsum[3];
    float gv = s / (float)(4 * P);
    if (!__builtin_isfinite(gv)) gv = 0.5f;
    gv = fminf(fmaxf(gv, 0.0f), 1.0f);
    int dk = (int)floorf(32.0f * gv);
    dk = dk < 1 ? 1 : (dk > 32 ? 32 : dk);
    scal[0] = (float)dk;
    scal[1] = a1[0] + a2[0] + a3[0] + a4[0];
  }
}

// dynamic top-k mask + softmax on 32-wide attention rows
__global__ __launch_bounds__(256) void attn_mask_softmax_k(
    float* __restrict__ attn, const float* __restrict__ temp,
    const float* __restrict__ scal)
{
  int idx = blockIdx.x * blockDim.x + threadIdx.x;
  if (idx >= 512) return;                 // 16 (b,h) * 32 rows
  int bh = idx >> 5, c = idx & 31, h = bh & 3;
  float tm = temp[h];
  float* row = attn + bh * 1024 + c * 32;
  float v[32];
  for (int d = 0; d < 32; ++d) v[d] = row[d] * tm;
  int dynk = (int)scal[0];
  int n = 32 - dynk;                      // index of threshold in ascending sort
  unsigned used = 0; float thr = v[0];
  for (int s = 0; s <= n; ++s) {
    float mn = 3.4e38f; int mi = 0;
    for (int d = 0; d < 32; ++d)
      if (!((used >> d) & 1u) && v[d] < mn) { mn = v[d]; mi = d; }
    used |= 1u << mi; thr = mn;
  }
  float mx = -3.4e38f;
  for (int d = 0; d < 32; ++d) if (v[d] >= thr) mx = fmaxf(mx, v[d]);
  float e[32], sum = 0.0f;
  for (int d = 0; d < 32; ++d) {
    e[d] = (v[d] >= thr) ? expf(v[d] - mx) : 0.0f;
    sum += e[d];
  }
  for (int d = 0; d < 32; ++d) row[d] = e[d] / sum;
}

__global__ __launch_bounds__(256) void scale_buf_k(float* buf, const float* scal, size_t total)
{
  size_t i = (size_t)blockIdx.x * 256 + threadIdx.x;
  if (i < total) buf[i] *= scal[1];
}

// channel-interaction MLP on pooled means (tiny, single block)
__global__ __launch_bounds__(128) void ci_k(
    const float* osum, const float* w1, const float* b1,
    const float* gng, const float* gnb,
    const float* w2, const float* b2, float* cm)
{
  __shared__ float t[32], g[32], mv[2];
  int tid = threadIdx.x;
  for (int b = 0; b < NB; ++b) {
    if (tid < 32) {
      float a = b1[tid];
      for (int c = 0; c < 128; ++c)
        a += w1[tid * 128 + c] * (osum[b * 128 + c] / (float)P);
      t[tid] = a;
    }
    __syncthreads();
    if (tid == 0) {
      float m = 0; for (int i = 0; i < 32; ++i) m += t[i]; m /= 32.0f;
      float v = 0; for (int i = 0; i < 32; ++i) { float d = t[i] - m; v += d * d; }
      mv[0] = m; mv[1] = v / 32.0f;
    }
    __syncthreads();
    if (tid < 32)
      g[tid] = gelu_exact((t[tid] - mv[0]) * rsqrtf(mv[1] + 1e-5f) * gng[tid] + gnb[tid]);
    __syncthreads();
    {
      float a = b2[tid];
      for (int i = 0; i < 32; ++i) a += w2[tid * 32 + i] * g[i];
      cm[b * 128 + tid] = a;
    }
    __syncthreads();
  }
}

// groupnorm (per-batch over 16*P) + gelu, in place on s1
__global__ __launch_bounds__(256) void gn_gelu_s1_k(
    float* s1, const float* rsum, const float* rsq,
    const float* g, const float* bb)
{
  size_t i = (size_t)blockIdx.x * 256 + threadIdx.x;
  if (i >= (size_t)NB * 16 * P) return;
  int b  = (int)(i / (16 * P));
  int ch = (int)((i / P) % 16);
  float cnt = (float)(16 * P);
  float m = rsum[b] / cnt;
  float v = rsq[b] / cnt - m * m;
  float x = s1[i];
  s1[i] = gelu_exact((x - m) * rsqrtf(v + 1e-5f) * g[ch] + bb[ch]);
}

// cat[:,0:128] = out_att * sigmoid(sm);  cat[:,128:256] = y_d * sigmoid(cm)
__global__ __launch_bounds__(256) void build_cat_k(
    const float* oatt, const float* yd, const float* sm,
    const float* cm, float* cat)
{
  size_t i = (size_t)blockIdx.x * 256 + threadIdx.x;
  if (i >= (size_t)NB * 128 * P) return;
  int  b = (int)(i / (128 * P));
  int  c = (int)((i / P) % 128);
  long p = (long)(i % P);
  float sv = sigm(sm[(size_t)b * P + p]);
  float cv = sigm(cm[b * 128 + c]);
  cat[(size_t)b * 256 * P + (size_t)c * P + p]         = oatt[i] * sv;
  cat[(size_t)b * 256 * P + (size_t)(128 + c) * P + p] = yd[i] * cv;
}

// DFT matrices: FC[k,n]=cos(2*pi*k*n/192), FS[k,n]=sin(2*pi*k*n/192)
__global__ __launch_bounds__(256) void init_dft_k(float* fc, float* fs)
{
  int i = blockIdx.x * 256 + threadIdx.x;
  if (i >= 192 * 192) return;
  int k = i / 192, n = i % 192;
  int t = (k * n) % 192;
  float ang = 6.28318530717958647692f * (float)t / 192.0f;
  fc[i] = cosf(ang);
  fs[i] = sinf(ang);
}

// channel-wise scale+bias on FFT real/imag
__global__ __launch_bounds__(256) void fft_scale_k(
    float* zr, float* zi, const float* fw, const float* fb)
{
  size_t i = (size_t)blockIdx.x * 256 + threadIdx.x;
  if (i >= (size_t)1024 * P) return;
  int c = (int)((i / P) % 256);
  zr[i] = zr[i] * fw[c]       + fb[c];
  zi[i] = zi[i] * fw[256 + c] + fb[256 + c];
}

// ---------------------------------------------------------------------------
// Host side
// ---------------------------------------------------------------------------
static inline void gemm(hipStream_t st,
    const float* A, int lda, long aso, long asi,
    const float* B, long bsk, long bsn, long bso, long bsi,
    float* D, int ldd, long dso, long dsi, const float* bias,
    int M, int N, int K, int Z, int ZH, float alpha, int accum, int act)
{
  dim3 g((N + 63) / 64, (M + 31) / 32, Z);
  gemm_bf16_wmma<<<g, 256, 0, st>>>(A, lda, aso, asi, B, bsk, bsn, bso, bsi,
                                    D, ldd, dso, dsi, bias,
                                    M, N, K, ZH, alpha, accum, act);
}
static inline dim3 ew(size_t n) { return dim3((unsigned)((n + 255) / 256)); }

// Workspace layout (floats).  Total ~352M floats (~1.41 GB).
static constexpr size_t SZ_BIG = 4UL * 256 * 36864;   // 37,748,736
static constexpr size_t SZ_YD  = 4UL * 128 * 36864;
static constexpr size_t SZ_QKV = 4UL * 384 * 36864;
static constexpr size_t F_LIN  = 0;
static constexpr size_t F_H1   = F_LIN  + SZ_BIG;
static constexpr size_t F_H2   = F_H1   + SZ_BIG;
static constexpr size_t F_YD   = F_H2   + SZ_BIG;
static constexpr size_t F_QKV  = F_YD   + SZ_YD;
static constexpr size_t F_QKVD = F_QKV  + SZ_QKV;
static constexpr size_t F_G1   = F_QKVD + SZ_QKV;
static constexpr size_t F_G2   = F_G1   + 4UL * 64 * 36864;
static constexpr size_t F_S1   = F_G2   + 4UL * 36864;
static constexpr size_t F_SM   = F_S1   + 4UL * 16 * 36864;
static constexpr size_t F_OATT = F_SM   + 4UL * 36864;
static constexpr size_t F_CAT  = F_OATT + SZ_YD;
static constexpr size_t F_Z    = F_CAT  + SZ_BIG;
static constexpr size_t F_FC   = F_Z    + SZ_BIG;
static constexpr size_t F_FS   = F_FC   + 36864;
static constexpr size_t F_RSUM = F_FS   + 36864;
static constexpr size_t F_RSQ  = F_RSUM + 2048;
static constexpr size_t F_SSOF = F_RSQ  + 2048;
static constexpr size_t F_SCAL = F_SSOF + 1024;
static constexpr size_t F_ATTN = F_SCAL + 16;
static constexpr size_t F_CM   = F_ATTN + 16384;

extern "C" void kernel_launch(void* const* d_in, const int* in_sizes, int n_in,
                              void* d_out, int out_size, void* d_ws, size_t ws_size,
                              hipStream_t stream)
{
  (void)in_sizes; (void)n_in; (void)out_size; (void)ws_size;
  const float* x       = (const float*)d_in[0];
  const float* lin0_w  = (const float*)d_in[1];
  const float* lin0_b  = (const float*)d_in[2];
  const float* sprdw_w = (const float*)d_in[3];
  const float* sprdw_b = (const float*)d_in[4];
  const float* sprpw_w = (const float*)d_in[5];
  const float* sprpw_b = (const float*)d_in[6];
  const float* sprout_w= (const float*)d_in[7];
  const float* sprout_b= (const float*)d_in[8];
  const float* qkv_w   = (const float*)d_in[9];
  const float* qkv_b   = (const float*)d_in[10];
  const float* qkvdw_w = (const float*)d_in[11];
  const float* qkvdw_b = (const float*)d_in[12];
  const float* temp    = (const float*)d_in[13];
  const float* a1      = (const float*)d_in[14];
  const float* a2      = (const float*)d_in[15];
  const float* a3      = (const float*)d_in[16];
  const float* a4      = (const float*)d_in[17];
  const float* ci_w1   = (const float*)d_in[18];
  const float* ci_b1   = (const float*)d_in[19];
  const float* ci_gng  = (const float*)d_in[20];
  const float* ci_gnb  = (const float*)d_in[21];
  const float* ci_w2   = (const float*)d_in[22];
  const float* ci_b2   = (const float*)d_in[23];
  const float* si_w1   = (const float*)d_in[24];
  const float* si_b1   = (const float*)d_in[25];
  const float* si_gng  = (const float*)d_in[26];
  const float* si_gnb  = (const float*)d_in[27];
  const float* si_w2   = (const float*)d_in[28];
  const float* si_b2   = (const float*)d_in[29];
  const float* gate_w1 = (const float*)d_in[30];
  const float* gate_b1 = (const float*)d_in[31];
  const float* gate_w2 = (const float*)d_in[32];
  const float* gate_b2 = (const float*)d_in[33];
  const float* proj_w  = (const float*)d_in[34];
  const float* proj_b  = (const float*)d_in[35];
  const float* fft_w   = (const float*)d_in[36];
  const float* fft_b   = (const float*)d_in[37];

  float* W     = (float*)d_ws;
  float* lin   = W + F_LIN;
  float* h1    = W + F_H1;
  float* h2    = W + F_H2;
  float* yd    = W + F_YD;
  float* qkv   = W + F_QKV;
  float* qkvd  = W + F_QKVD;
  float* g1    = W + F_G1;
  float* g2    = W + F_G2;
  float* s1    = W + F_S1;
  float* smb   = W + F_SM;
  float* oatt  = W + F_OATT;
  float* cat   = W + F_CAT;
  float* zbuf  = W + F_Z;
  float* fc    = W + F_FC;
  float* fsm   = W + F_FS;
  float* rsum  = W + F_RSUM;
  float* rsq   = W + F_RSQ;
  float* ssoft = W + F_SSOF;
  float* scal  = W + F_SCAL;
  float* attn  = W + F_ATTN;
  float* cm    = W + F_CM;
  float* outp  = (float*)d_out;
  const long IMG = 36864;

  init_dft_k<<<144, 256, 0, stream>>>(fc, fsm);

  // lin = lin0(x)                                   [B,256,P]
  gemm(stream, lin0_w, 256, 0, 0,  x, P, 1, 256 * P, 0,
       lin, (int)P, 256 * P, 0, lin0_b, 256, (int)P, 256, NB, 1, 1.f, 0, 0);

  // spr_sa depthwise (y = lin[:, :128], groups=128 -> 256 ch)
  dwconv3x3_k<<<ew(4UL * 256 * P), 256, 0, stream>>>(lin, 256 * P, sprdw_w, sprdw_b, h1, 256, 2);
  // spr pointwise 256->256
  gemm(stream, sprpw_w, 256, 0, 0,  h1, P, 1, 256 * P, 0,
       h2, (int)P, 256 * P, 0, sprpw_b, 256, (int)P, 256, NB, 1, 1.f, 0, 0);
  // channel softmax of spatial means, then gelu(s*h)
  row_reduce_k<<<1024, 256, 0, stream>>>(h2, 256 * P, P, 256, P, rsum, rsq);
  spr_softmax_k<<<NB, 256, 0, stream>>>(rsum, ssoft);
  scale_gelu_k<<<ew(4UL * 256 * P), 256, 0, stream>>>(h2, ssoft);
  // spr out 256->128  -> y_d
  gemm(stream, sprout_w, 256, 0, 0,  h2, P, 1, 256 * P, 0,
       yd, (int)P, 128 * P, 0, sprout_b, 128, (int)P, 256, NB, 1, 1.f, 0, 0);

  // qkv = conv(xs) 128->384, then depthwise 3x3 (groups=384)
  gemm(stream, qkv_w, 128, 0, 0,  lin + 128 * P, P, 1, 256 * P, 0,
       qkv, (int)P, 384 * P, 0, qkv_b, 384, (int)P, 128, NB, 1, 1.f, 0, 0);
  dwconv3x3_k<<<ew(4UL * 384 * P), 256, 0, stream>>>(qkv, 384 * P, qkvdw_w, qkvdw_b, qkvd, 384, 1);

  // gate: 128->64 relu, 64->1 sigmoid, mean -> dyn_k
  gemm(stream, gate_w1, 128, 0, 0,  lin + 128 * P, P, 1, 256 * P, 0,
       g1, (int)P, 64 * P, 0, gate_b1, 64, (int)P, 128, NB, 1, 1.f, 0, 1);
  gemm(stream, gate_w2, 64, 0, 0,  g1, P, 1, 64 * P, 0,
       g2, (int)P, P, 0, gate_b2, 1, (int)P, 64, NB, 1, 1.f, 0, 2);
  row_reduce_k<<<NB, 256, 0, stream>>>(g2, P, P, 1, P, rsum, rsq);
  gv_dynk_k<<<1, 32, 0, stream>>>(rsum, a1, a2, a3, a4, scal);

  // q,k L2 normalize along n
  row_reduce_k<<<1024, 256, 0, stream>>>(qkvd, 384 * P, P, 256, P, rsum, rsq);
  qk_norm_k<<<ew(4UL * 256 * P), 256, 0, stream>>>(qkvd, rsq);

  // attn = q @ k^T  (per b,h; K = 36864)
  gemm(stream, qkvd, (int)P, 384 * P, 32 * P,
       qkvd + 128 * P, 1, P, 384 * P, 32 * P,
       attn, 32, 4096, 1024, nullptr, 32, 32, (int)P, 16, 4, 1.f, 0, 0);
  attn_mask_softmax_k<<<2, 256, 0, stream>>>(attn, temp, scal);
  // out = attn @ v
  gemm(stream, attn, 32, 4096, 1024,
       qkvd + 256 * P, P, 1, 384 * P, 32 * P,
       oatt, (int)P, 128 * P, 32 * P, nullptr, 32, (int)P, 32, 16, 4, 1.f, 0, 0);
  scale_buf_k<<<ew(4UL * 128 * P), 256, 0, stream>>>(oatt, scal, (size_t)4 * 128 * P);

  // channel interaction (pooled means -> cm)
  row_reduce_k<<<512, 256, 0, stream>>>(oatt, 128 * P, P, 128, P, rsum, rsq);
  ci_k<<<1, 128, 0, stream>>>(rsum, ci_w1, ci_b1, ci_gng, ci_gnb, ci_w2, ci_b2, cm);

  // spatial interaction: si1 -> gn+gelu -> si2
  gemm(stream, si_w1, 128, 0, 0,  yd, P, 1, 128 * P, 0,
       s1, (int)P, 16 * P, 0, si_b1, 16, (int)P, 128, NB, 1, 1.f, 0, 0);
  row_reduce_k<<<NB, 256, 0, stream>>>(s1, 16 * P, P, 1, 16 * P, rsum, rsq);
  gn_gelu_s1_k<<<ew(4UL * 16 * P), 256, 0, stream>>>(s1, rsum, rsq, si_gng, si_gnb);
  gemm(stream, si_w2, 16, 0, 0,  s1, P, 1, 16 * P, 0,
       smb, (int)P, P, 0, si_b2, 1, (int)P, 16, NB, 1, 1.f, 0, 0);

  // concat gated branches, then proj 256->256 -> z
  build_cat_k<<<ew(4UL * 128 * P), 256, 0, stream>>>(oatt, yd, smb, cm, cat);
  gemm(stream, proj_w, 256, 0, 0,  cat, P, 1, 256 * P, 0,
       zbuf, (int)P, 256 * P, 0, proj_b, 256, (int)P, 256, NB, 1, 1.f, 0, 0);

  // -------- FFT2 as DFT GEMMs (F = FC - i*FS) over 1024 (b,c) images -------
  float* Yr = lin;  float* Yi = h1;      // dead buffers reused
  float* Zr = qkv;  float* Zi = qkvd;
  float* Br = h2;   float* Bi = cat;
  // forward stage 1 (along w):  Yr = FC@X, Yi = -FS@X   (Y stored [kw][h])
  gemm(stream, fc, 192, 0, 0,  zbuf, 1, 192, IMG, 0, Yr, 192, IMG, 0, nullptr, 192, 192, 192, 1024, 1,  1.f, 0, 0);
  gemm(stream, fsm, 192, 0, 0, zbuf, 1, 192, IMG, 0, Yi, 192, IMG, 0, nullptr, 192, 192, 192, 1024, 1, -1.f, 0, 0);
  // forward stage 2 (along h): Zr = FC@Yr + FS@Yi ; Zi = FC@Yi - FS@Yr
  gemm(stream, fc, 192, 0, 0,  Yr, 1, 192, IMG, 0, Zr, 192, IMG, 0, nullptr, 192, 192, 192, 1024, 1,  1.f, 0, 0);
  gemm(stream, fsm, 192, 0, 0, Yi, 1, 192, IMG, 0, Zr, 192, IMG, 0, nullptr, 192, 192, 192, 1024, 1,  1.f, 1, 0);
  gemm(stream, fc, 192, 0, 0,  Yi, 1, 192, IMG, 0, Zi, 192, IMG, 0, nullptr, 192, 192, 192, 1024, 1,  1.f, 0, 0);
  gemm(stream, fsm, 192, 0, 0, Yr, 1, 192, IMG, 0, Zi, 192, IMG, 0, nullptr, 192, 192, 192, 1024, 1, -1.f, 1, 0);
  // per-channel scale/bias on real & imag
  fft_scale_k<<<ew(1024UL * P), 256, 0, stream>>>(Zr, Zi, fft_w, fft_b);
  // inverse stage 1: Br = FC@Zr - FS@Zi ; Bi = FC@Zi + FS@Zr
  gemm(stream, fc, 192, 0, 0,  Zr, 1, 192, IMG, 0, Br, 192, IMG, 0, nullptr, 192, 192, 192, 1024, 1,  1.f, 0, 0);
  gemm(stream, fsm, 192, 0, 0, Zi, 1, 192, IMG, 0, Br, 192, IMG, 0, nullptr, 192, 192, 192, 1024, 1, -1.f, 1, 0);
  gemm(stream, fc, 192, 0, 0,  Zi, 1, 192, IMG, 0, Bi, 192, IMG, 0, nullptr, 192, 192, 192, 1024, 1,  1.f, 0, 0);
  gemm(stream, fsm, 192, 0, 0, Zr, 1, 192, IMG, 0, Bi, 192, IMG, 0, nullptr, 192, 192, 192, 1024, 1,  1.f, 1, 0);
  // inverse stage 2, real part only, with 1/(192*192) normalization
  const float inv = 1.0f / 36864.0f;
  gemm(stream, fc, 192, 0, 0,  Br, 1, 192, IMG, 0, outp, 192, IMG, 0, nullptr, 192, 192, 192, 1024, 1,  inv, 0, 0);
  gemm(stream, fsm, 192, 0, 0, Bi, 1, 192, IMG, 0, outp, 192, IMG, 0, nullptr, 192, 192, 192, 1024, 1, -inv, 1, 0);
}
